// MultiHeadClassifier_78554951843974
// MI455X (gfx1250) — matrix operand order repaired and checked by reading, not compiled, not run on hardware
//
#include <hip/hip_runtime.h>
#include <hip/hip_bf16.h>

// ---------------- problem constants ----------------
#define N_TASKS   8
#define D_MODEL   1024
#define HIDDEN    1024
#define N_CLASSES 100
#define BATCH     8192

// ---------------- tiling ----------------
#define BM     64            // rows per segment/block
#define BN1    256           // layer-1 N tile (8 waves = 2M x 4N, 32x64 per wave)
#define KST    32            // K step (WMMA bf16 K)
#define XPITCH 1032          // LDS row pitch (bf16 elems), +8 pad -> conflict-free b128
#define BPITCH 40            // LDS B-tile row pitch (bf16 elems), 80B, 16B aligned
#define BROWS1 256           // layer-1 B-tile rows
#define BROWS2 128           // layer-2 B-tile rows (100 -> 128 padded)
#define NSEG   (BATCH / BM + N_TASKS)    // 136
#define PADDED (BATCH + N_TASKS * BM)    // 8704

// ---------------- workspace layout ----------------
// int region (word offsets)
#define WS_COUNTS 0
#define WS_OFFS   8
#define WS_CURS   16
#define WS_SEGT   64
#define WS_SEGB   256
#define WS_ROWIDX 512
// byte offsets for bf16 staging areas
#define WS_XBF_OFF  65536
#define WS_XBF_B    (BATCH * D_MODEL * 2)              // 16 MB
#define WS_W1T_OFF  (WS_XBF_OFF + WS_XBF_B)
#define WS_W1T_B    (N_TASKS * HIDDEN * D_MODEL * 2)   // 16 MB  [t][n][k]
#define WS_W2T_OFF  (WS_W1T_OFF + WS_W1T_B)
#define WS_W2T_B    (N_TASKS * 128 * HIDDEN * 2)       // 2 MB   [t][n(128)][k]

// ---------------- LDS layout (bytes) ----------------
#define SMEM_ROWS_B 256
#define SMEM_X_OFF  256
#define SMEM_TILE_B (BM * XPITCH * 2)            // 132096
#define SMEM_H_OFF  (SMEM_X_OFF + SMEM_TILE_B)   // 132352
#define SMEM_B_OFF  (SMEM_H_OFF + SMEM_TILE_B)   // 264448
#define SMEM_B_B    (BROWS1 * BPITCH * 2)        // 20480 per buffer
#define SMEM_BYTES  (SMEM_B_OFF + 2 * SMEM_B_B)  // 305408 < 320K per WGP

typedef __attribute__((ext_vector_type(16))) __bf16 v16bf;
typedef __attribute__((ext_vector_type(8)))  __bf16 v8bf;
typedef __attribute__((ext_vector_type(4)))  __bf16 v4bf;
typedef __attribute__((ext_vector_type(8)))  float  v8f;

__device__ __forceinline__ __bf16 f2bf(float f) { return (__bf16)f; }

// ---- gfx1250 async copy to LDS (ASYNCcnt-tracked, VGPR bypass) ----
__device__ __forceinline__ void async_b128(unsigned lds_off, unsigned long long gaddr) {
    asm volatile("global_load_async_to_lds_b128 %0, %1, off"
                 :: "v"(lds_off), "v"(gaddr) : "memory");
}
template <int N>
__device__ __forceinline__ void wait_asynccnt() {
    asm volatile("s_wait_asynccnt %0" :: "n"(N) : "memory");
}

// A fragment (16x32 bf16) per ISA layout:
// lane l, elem i: m = (l&15), k = (i&7) + ((l>>4)<<3) + ((i>>3)<<4)
__device__ __forceinline__ v16bf load_a_frag(const __bf16* lds, int m_base, int k_base, int lane) {
    const __bf16* p = lds + (m_base + (lane & 15)) * XPITCH + k_base + ((lane >> 4) << 3);
    v8bf lo = *(const v8bf*)p;
    v8bf hi = *(const v8bf*)(p + 16);
    return __builtin_shufflevector(lo, hi, 0,1,2,3,4,5,6,7,8,9,10,11,12,13,14,15);
}

// B fragment (32x16 bf16), tile staged transposed in LDS as [n][k]:
// lane l, elem i: n = (l&15), k = (i&7) + ((l>>4)<<3) + ((i>>3)<<4)
__device__ __forceinline__ v16bf load_b_frag(const __bf16* blds, int n_base, int lane) {
    const __bf16* p = blds + (n_base + (lane & 15)) * BPITCH + ((lane >> 4) << 3);
    v8bf lo = *(const v8bf*)p;
    v8bf hi = *(const v8bf*)(p + 16);
    return __builtin_shufflevector(lo, hi, 0,1,2,3,4,5,6,7,8,9,10,11,12,13,14,15);
}

// stage an NROWS x 64B tile (rows stride 1024 bf16) into LDS via async copies.
// NROWS*4 16B chunks -> NROWS/64 issues per thread.
template <int NROWS>
__device__ __forceinline__ void stage_b_async(const __bf16* src_base, unsigned lds_base, int tid) {
    #pragma unroll
    for (int it = 0; it < (NROWS * 4) / 256; ++it) {
        int idx = tid + it * 256;
        int n = idx >> 2, c = idx & 3;
        unsigned long long ga =
            (unsigned long long)(uintptr_t)(src_base + (size_t)n * D_MODEL) + (unsigned)(c * 16);
        async_b128(lds_base + (unsigned)(n * (BPITCH * 2) + c * 16), ga);
    }
}

// ---------------- bucketing kernels ----------------
__global__ void k_init(int* ws) {
    int i = blockIdx.x * 256 + threadIdx.x;
    if (i < 8)     { ws[WS_COUNTS + i] = 0; ws[WS_CURS + i] = 0; }
    if (i < NSEG)  { ws[WS_SEGT + i] = -1; ws[WS_SEGB + i] = 0; }
    if (i < PADDED) ws[WS_ROWIDX + i] = -1;
}

__global__ void k_count(const int* __restrict__ task_id, int* ws) {
    int b = blockIdx.x * 256 + threadIdx.x;
    if (b < BATCH) atomicAdd(&ws[WS_COUNTS + task_id[b]], 1);
}

__global__ void k_plan(int* ws) {
    int off = 0, s = 0;
    for (int t = 0; t < N_TASKS; ++t) {
        ws[WS_OFFS + t] = off;
        int c  = ws[WS_COUNTS + t];
        int ns = (c + BM - 1) / BM;
        for (int i = 0; i < ns; ++i) { ws[WS_SEGT + s] = t; ws[WS_SEGB + s] = off + i * BM; ++s; }
        off += ns * BM;
    }
}

__global__ void k_scatter(const int* __restrict__ task_id, int* ws) {
    int b = blockIdx.x * 256 + threadIdx.x;
    if (b < BATCH) {
        int t   = task_id[b];
        int pos = atomicAdd(&ws[WS_CURS + t], 1);
        ws[WS_ROWIDX + ws[WS_OFFS + t] + pos] = b;
    }
}

// ---------------- precision/layout prep kernels ----------------
__global__ void k_cvt_x(const float4* __restrict__ x, __bf16* __restrict__ xbf) {
    int idx = blockIdx.x * 256 + threadIdx.x;   // over BATCH*D/4
    float4 v = x[idx];
    v4bf o; o.x = f2bf(v.x); o.y = f2bf(v.y); o.z = f2bf(v.z); o.w = f2bf(v.w);
    *(v4bf*)(xbf + (size_t)idx * 4) = o;
}

// W1 [t][k][n] f32 -> [t][n][k] bf16 (LDS-tiled transpose, coalesced both sides)
__global__ void k_cvt_w1(const float* __restrict__ W1, __bf16* __restrict__ w1t) {
    __shared__ float tile[32][33];
    int t  = blockIdx.z;
    int k0 = blockIdx.y * 32;
    int n0 = blockIdx.x * 32;
    int tx = threadIdx.x & 31, ty = threadIdx.x >> 5;   // 256 threads
    const float* src = W1 + ((size_t)t * D_MODEL + k0) * HIDDEN + n0;
    #pragma unroll
    for (int i = 0; i < 4; ++i) tile[ty + 8 * i][tx] = src[(ty + 8 * i) * HIDDEN + tx];
    __syncthreads();
    __bf16* dst = w1t + ((size_t)t * HIDDEN + n0) * D_MODEL + k0;
    #pragma unroll
    for (int i = 0; i < 4; ++i) dst[(size_t)(ty + 8 * i) * D_MODEL + tx] = f2bf(tile[tx][ty + 8 * i]);
}

// W2 [t][k][100] f32 -> [t][n(128 zero-padded)][k] bf16
__global__ void k_cvt_w2(const float* __restrict__ W2, __bf16* __restrict__ w2t) {
    int idx = blockIdx.x * 256 + threadIdx.x;   // over 8*128*1024
    int k = idx & 1023;
    int n = (idx >> 10) & 127;
    int t = idx >> 17;
    float v = (n < N_CLASSES) ? W2[((size_t)t * D_MODEL + k) * N_CLASSES + n] : 0.f;
    w2t[idx] = f2bf(v);
}

// ---------------- fused per-segment MLP ----------------
__global__ __launch_bounds__(256, 1)
void k_mlp(const __bf16* __restrict__ xbf, const __bf16* __restrict__ w1t,
           const __bf16* __restrict__ w2t, const float* __restrict__ b1,
           const float* __restrict__ b2,   const int*   __restrict__ ws,
           float* __restrict__ out) {
    extern __shared__ __align__(16) char smem[];
    int*    rows_s = (int*)smem;
    __bf16* Xlds   = (__bf16*)(smem + SMEM_X_OFF);
    __bf16* Hlds   = (__bf16*)(smem + SMEM_H_OFF);
    __bf16* Blds   = (__bf16*)(smem + SMEM_B_OFF);

    const int seg  = blockIdx.x;
    const int task = ws[WS_SEGT + seg];
    if (task < 0) return;
    const int base = ws[WS_SEGB + seg];

    const int tid  = threadIdx.x;
    const int lane = tid & 31;
    const int wave = tid >> 5;

    if (tid < BM) rows_s[tid] = ws[WS_ROWIDX + base + tid];
    __syncthreads();

    // ---- gather X tile (bf16) into LDS via async copies; zero invalid rows ----
    #pragma unroll 1
    for (int it = 0; it < 32; ++it) {
        int idx = tid + it * 256;      // 64 rows x 128 16B-chunks
        int m = idx >> 7, c = idx & 127;
        int r = rows_s[m];
        unsigned ldso = (unsigned)(SMEM_X_OFF + m * (XPITCH * 2) + c * 16);
        if (r >= 0) {
            unsigned long long ga =
                (unsigned long long)(uintptr_t)(xbf + (size_t)r * D_MODEL) + (unsigned)(c * 16);
            async_b128(ldso, ga);
        } else {
            v8bf z = {};
            *(v8bf*)(smem + ldso) = z;
        }
    }

    // ================= layer 1: H = relu(X @ W1[t] + b1[t]) =================
    const int wm = wave >> 2;   // 0..1 -> 32 rows
    const int wn = wave & 3;    // 0..3 -> 64 cols of the 256-wide N tile
    const float* b1g = b1 + (size_t)task * HIDDEN;
    const __bf16* w1base = w1t + (size_t)task * HIDDEN * D_MODEL;

    #pragma unroll 1
    for (int n0 = 0; n0 < HIDDEN; n0 += BN1) {   // 4 passes
        v8f acc[2][4];
        #pragma unroll
        for (int mf = 0; mf < 2; ++mf)
            #pragma unroll
            for (int nf = 0; nf < 4; ++nf) acc[mf][nf] = (v8f){};
        // prologue: stage k=0 tile into buffer 0
        stage_b_async<BROWS1>(w1base + (size_t)n0 * D_MODEL, SMEM_B_OFF, tid);
        int cur = 0;
        #pragma unroll 1
        for (int k0 = 0; k0 < D_MODEL; k0 += KST) {
            wait_asynccnt<0>();   // current tile (and X on first pass) landed
            __syncthreads();      // visible to all; everyone done with buf^1
            if (k0 + KST < D_MODEL)
                stage_b_async<BROWS1>(w1base + (size_t)n0 * D_MODEL + (k0 + KST),
                                      SMEM_B_OFF + (unsigned)((cur ^ 1) * SMEM_B_B), tid);
            const __bf16* BB = Blds + cur * (BROWS1 * BPITCH);
            v16bf a0 = load_a_frag(Xlds, wm * 32,      k0, lane);
            v16bf a1 = load_a_frag(Xlds, wm * 32 + 16, k0, lane);
            v16bf bfr[4];
            #pragma unroll
            for (int nf = 0; nf < 4; ++nf) bfr[nf] = load_b_frag(BB, wn * 64 + nf * 16, lane);
            #pragma unroll
            for (int nf = 0; nf < 4; ++nf) {
                acc[0][nf] = __builtin_amdgcn_wmma_f32_16x16x32_bf16(false, a0, false, bfr[nf], (short)0, acc[0][nf], false, false);
                acc[1][nf] = __builtin_amdgcn_wmma_f32_16x16x32_bf16(false, a1, false, bfr[nf], (short)0, acc[1][nf], false, false);
            }
            cur ^= 1;
        }
        // epilogue: bias + relu -> Hlds (bf16). C layout: m = v + 8*(lane>=16), n = lane&15
        #pragma unroll
        for (int mf = 0; mf < 2; ++mf) {
            #pragma unroll
            for (int nf = 0; nf < 4; ++nf) {
                int n    = n0 + wn * 64 + nf * 16 + (lane & 15);
                float bs = b1g[n];
                int mrow = wm * 32 + mf * 16 + ((lane >> 4) << 3);
                #pragma unroll
                for (int v = 0; v < 8; ++v) {
                    float val = acc[mf][nf][v] + bs;
                    Hlds[(mrow + v) * XPITCH + n] = f2bf(fmaxf(val, 0.f));
                }
            }
        }
    }
    __syncthreads();   // Hlds complete; Blds buffers free

    // ================= layer 2: OUT = H @ W2[t] + b2[t] (128 padded cols) =================
    const int wm2    = wave & 3;        // m-frag (16 rows)
    const int nfbase = (wave >> 2) * 4; // nfrags 0..3 / 4..7 (7 = padding)
    const float* b2g = b2 + (size_t)task * N_CLASSES;
    const __bf16* w2base = w2t + (size_t)task * 128 * D_MODEL;
    v8f acc2[4];
    #pragma unroll
    for (int j = 0; j < 4; ++j) acc2[j] = (v8f){};

    stage_b_async<BROWS2>(w2base, SMEM_B_OFF, tid);
    int cur2 = 0;
    #pragma unroll 1
    for (int k0 = 0; k0 < HIDDEN; k0 += KST) {
        wait_asynccnt<0>();
        __syncthreads();
        if (k0 + KST < HIDDEN)
            stage_b_async<BROWS2>(w2base + (k0 + KST),
                                  SMEM_B_OFF + (unsigned)((cur2 ^ 1) * SMEM_B_B), tid);
        const __bf16* BB = Blds + cur2 * (BROWS1 * BPITCH);
        v16bf a = load_a_frag(Hlds, wm2 * 16, k0, lane);
        #pragma unroll
        for (int j = 0; j < 4; ++j) {
            v16bf bb = load_b_frag(BB, (nfbase + j) * 16, lane);
            acc2[j] = __builtin_amdgcn_wmma_f32_16x16x32_bf16(false, a, false, bb, (short)0, acc2[j], false, false);
        }
        cur2 ^= 1;
    }

    // epilogue: bias + scatter valid rows/cols
    #pragma unroll
    for (int j = 0; j < 4; ++j) {
        int n = (nfbase + j) * 16 + (lane & 15);
        if (n < N_CLASSES) {
            float bs  = b2g[n];
            int mbase = wm2 * 16 + ((lane >> 4) << 3);
            #pragma unroll
            for (int v = 0; v < 8; ++v) {
                int r = rows_s[mbase + v];
                if (r >= 0) out[(size_t)r * N_CLASSES + n] = acc2[j][v] + bs;
            }
        }
    }
}

// ---------------- launch ----------------
extern "C" void kernel_launch(void* const* d_in, const int* in_sizes, int n_in,
                              void* d_out, int out_size, void* d_ws, size_t ws_size,
                              hipStream_t stream) {
    (void)in_sizes; (void)n_in; (void)out_size; (void)ws_size;
    const float* x       = (const float*)d_in[0];
    const int*   task_id = (const int*)  d_in[1];
    const float* W1      = (const float*)d_in[2];
    const float* b1      = (const float*)d_in[3];
    const float* W2      = (const float*)d_in[4];
    const float* b2      = (const float*)d_in[5];
    float*       out     = (float*)d_out;
    int*         ws      = (int*)d_ws;
    __bf16*      xbf     = (__bf16*)((char*)d_ws + WS_XBF_OFF);
    __bf16*      w1t     = (__bf16*)((char*)d_ws + WS_W1T_OFF);
    __bf16*      w2t     = (__bf16*)((char*)d_ws + WS_W2T_OFF);

    (void)hipFuncSetAttribute((const void*)k_mlp,
                              hipFuncAttributeMaxDynamicSharedMemorySize,
                              (int)SMEM_BYTES);

    k_init   <<<(PADDED + 255) / 256, 256, 0, stream>>>(ws);
    k_count  <<<BATCH / 256,          256, 0, stream>>>(task_id, ws);
    k_plan   <<<1, 1, 0, stream>>>(ws);
    k_scatter<<<BATCH / 256,          256, 0, stream>>>(task_id, ws);
    k_cvt_x  <<<(BATCH * D_MODEL / 4) / 256, 256, 0, stream>>>((const float4*)x, xbf);
    k_cvt_w1 <<<dim3(32, 32, 8), 256, 0, stream>>>(W1, w1t);
    k_cvt_w2 <<<(N_TASKS * 128 * HIDDEN) / 256, 256, 0, stream>>>(W2, w2t);
    k_mlp    <<<NSEG, 256, SMEM_BYTES, stream>>>(xbf, w1t, w2t, b1, b2, ws, out);
}
